// ModernDenseRecon_75376676045348
// MI455X (gfx1250) — compile-verified, matrix operand order checked by reading, not run on hardware
//
#include <hip/hip_runtime.h>

#define BATCH 4096
#define DEPTH 1024
#define NMEM  16384
#define NB    32             // memory slots per iteration (full K=32 retrieval WMMA)
#define ROWS  32             // batch rows per workgroup (2 x 16-row sub-tiles)
#define THREADS 512          // 16 wave32 per workgroup; each wave owns 64 depth cols
#define WPITCH 1040          // row-major W block pitch (bf16): 512 + 8 pad + 512 + 8 pad
#define TPITCH 40            // transposed W block pitch (bf16 elems), 80B, 16B-aligned
#define SPITCH 36            // score tile pitch (f32), 144B, 16B-aligned

typedef __bf16 v16bf __attribute__((ext_vector_type(16)));
typedef float  v8f   __attribute__((ext_vector_type(8)));
typedef unsigned int tdm_v4u __attribute__((ext_vector_type(4)));
typedef int          tdm_v8i __attribute__((ext_vector_type(8)));
typedef int          tdm_v4i __attribute__((ext_vector_type(4)));

#define HAVE_TDM __has_builtin(__builtin_amdgcn_tensor_load_to_lds)

__device__ __forceinline__ unsigned short f32_to_bf16_bits(float f) {
    unsigned int u = __builtin_bit_cast(unsigned int, f);
    unsigned int r = u + 0x7FFFu + ((u >> 16) & 1u);   // round-to-nearest-even
    return (unsigned short)(r >> 16);
}
__device__ __forceinline__ __bf16 bits_to_bf16(unsigned short s) {
    return __builtin_bit_cast(__bf16, s);
}
__device__ __forceinline__ unsigned int pack_bf16x2(float a, float b) {
    return (unsigned int)f32_to_bf16_bits(a) | ((unsigned int)f32_to_bf16_bits(b) << 16);
}
// Wlds element index with TDM mid-row padding (8 elems after every 512)
__device__ __forceinline__ int widx(int r, int c) {
    return r * WPITCH + c + ((c >> 9) << 3);
}

// ---------------- prepass: W f32 -> bf16, row-major + block-transposed ----------------
__global__ __launch_bounds__(256)
void convert_w_bf16(const float* __restrict__ W,
                    unsigned int* __restrict__ W16,
                    unsigned int* __restrict__ W16T) {
    const size_t t = (size_t)blockIdx.x * 256 + threadIdx.x;   // 0 .. 8,388,607 pairs
    {   // row-major pair
        const float2 f2 = *(const float2*)(W + t * 2);
        W16[t] = pack_bf16x2(f2.x, f2.y);
    }
    {   // block-transposed pair: t -> (nb, d, n2)
        const int d  = (int)(t & 1023);
        const int n2 = (int)((t >> 10) & 15);
        const int nb = (int)(t >> 14);
        const int n  = n2 * 2;
        const float a = W[(size_t)(nb * NB + n)     * DEPTH + d];
        const float b = W[(size_t)(nb * NB + n + 1) * DEPTH + d];
        W16T[(size_t)nb * (DEPTH * NB / 2) + d * (NB / 2) + n2] = pack_bf16x2(a, b);
    }
}

// ---------------- TDM descriptor helper (2D tile, bf16, LDS padding) ----------------
#if HAVE_TDM
__device__ __forceinline__ void tdm_load_2d(unsigned long long gaddr, unsigned lds_addr,
                                            int dim0, int dim1, int stride0,
                                            int pad_ivl, int pad_amt) {
    tdm_v4u g0;
    g0[0] = 1u;                                            // count=1, user descriptor
    g0[1] = lds_addr;                                      // LDS byte address
    g0[2] = (unsigned)(gaddr & 0xFFFFFFFFull);             // global_addr[31:0]
    g0[3] = (unsigned)((gaddr >> 32) & 0x1FFFFFFull)       // global_addr[56:32]
          | 0x80000000u;                                   // type=2 ("image")
    tdm_v8i g1;
    g1[0] = (1 << 16)                                      // data_size = 2 bytes
          | (1 << 20)                                      // pad_enable
          | (pad_ivl << 22) | (pad_amt << 25);
    g1[1] = (dim0 & 0xFFFF) << 16;                         // tensor_dim0[15:0]
    g1[2] = ((dim0 >> 16) & 0xFFFF) | ((dim1 & 0xFFFF) << 16);
    g1[3] = ((dim1 >> 16) & 0xFFFF) | ((dim0 & 0xFFFF) << 16);  // tile_dim0 = dim0
    g1[4] = (dim1 & 0xFFFF);                               // tile_dim1 = dim1, tile_dim2=0
    g1[5] = stride0;                                       // tensor_dim0_stride[31:0]
    g1[6] = 0;
    g1[7] = 0;
    tdm_v4i gz = {0, 0, 0, 0};
#if __clang_major__ >= 23
    tdm_v8i gz8 = {0, 0, 0, 0, 0, 0, 0, 0};
    __builtin_amdgcn_tensor_load_to_lds(g0, g1, gz, gz, gz8, 0);
#else
    __builtin_amdgcn_tensor_load_to_lds(g0, g1, gz, gz, 0);
#endif
}
#endif

// ---------------- fused flash-Hopfield kernel ----------------
#define WLDS_ELEMS   (NB * WPITCH)
#define WLDST_ELEMS  (DEPTH * TPITCH)
#define SFULL_FLOATS (2 * 2 * 16 * SPITCH)
#define SMEM_BYTES (WLDS_ELEMS * 2 + WLDST_ELEMS * 2 + SFULL_FLOATS * 4)

extern __shared__ __align__(16) unsigned char smem_raw[];

__global__ __launch_bounds__(THREADS)
void hopfield_flash_wmma(const float* __restrict__ x,
                         const float* __restrict__ W,
                         const unsigned int* __restrict__ W16,
                         const unsigned int* __restrict__ W16T,
                         float* __restrict__ out,
                         int use_bf16_src) {
    unsigned short* Wlds  = (unsigned short*)smem_raw;
    unsigned short* WldsT = Wlds + WLDS_ELEMS;
    float*          Sfull = (float*)(WldsT + WLDST_ELEMS);

    const int tid  = threadIdx.x;
    const int lane = tid & 31;
    const int wv   = tid >> 5;          // wave 0..15
    const int hi   = lane >> 4;         // half-wave
    const int ln16 = lane & 15;
    const int koff = hi * 8;
    const int rowbase = blockIdx.x * ROWS;
    const int dw   = wv * 64;           // this wave's depth slice

    // ---- A fragments: x tile, bf16, kept in VGPRs for the whole loop ----
    v16bf af[2][2];
    #pragma unroll
    for (int rt = 0; rt < 2; ++rt) {
        const float* xr = x + (size_t)(rowbase + rt * 16 + ln16) * DEPTH;
        #pragma unroll
        for (int c = 0; c < 2; ++c) {
            const int k0 = dw + 32 * c;
            #pragma unroll
            for (int i = 0; i < 8; ++i)
                af[rt][c][i]     = bits_to_bf16(f32_to_bf16_bits(xr[k0 + koff + i]));
            #pragma unroll
            for (int i = 0; i < 8; ++i)
                af[rt][c][8 + i] = bits_to_bf16(f32_to_bf16_bits(xr[k0 + 16 + koff + i]));
        }
    }

    v8f accO[2][4];
    #pragma unroll
    for (int rt = 0; rt < 2; ++rt)
        #pragma unroll
        for (int t = 0; t < 4; ++t)
            #pragma unroll
            for (int v = 0; v < 8; ++v) accO[rt][t][v] = 0.0f;
    float mrow[2] = {-3.0e38f, -3.0e38f};
    float lrow[2] = {0.0f, 0.0f};

    const int NBLK = NMEM / NB;   // 512
    #pragma unroll 1
    for (int nb = 0; nb < NBLK; ++nb) {
        __syncthreads();          // (A) previous Wlds/WldsT fully consumed

        const unsigned int* w16blk  = W16  + (size_t)nb * (NB * DEPTH / 2);
        const unsigned int* w16tblk = W16T + (size_t)nb * (DEPTH * NB / 2);
        const float*        wb      = W    + (size_t)nb * NB * DEPTH;

        // prefetch next block into GL2 (global_prefetch_b8)
        if (nb + 1 < NBLK) {
            if (use_bf16_src) {
                __builtin_prefetch((const char*)(w16blk  + NB * DEPTH / 2) + tid * 128, 0, 1);
                __builtin_prefetch((const char*)(w16tblk + DEPTH * NB / 2) + tid * 128, 0, 1);
            } else {
                __builtin_prefetch(wb + NB * DEPTH + (size_t)tid * 64, 0, 1);
                __builtin_prefetch(wb + NB * DEPTH + (size_t)tid * 64 + 32, 0, 1);
            }
        }

        // ---- stage W block into LDS ----
        bool manual_stage = true;
#if HAVE_TDM
        if (use_bf16_src) {
            if (wv == 0) {
                // row-major [32 x 1024] bf16; pad 4 DW after every 256 DW -> pitch 1040
                tdm_load_2d((unsigned long long)(const void*)w16blk,
                            (unsigned)(unsigned long long)(uintptr_t)Wlds,
                            DEPTH, NB, DEPTH, /*pad_ivl=*/7, /*pad_amt=*/3);
                // transposed [1024 x 32] bf16; pad 4 DW after every 16 DW -> pitch 40
                tdm_load_2d((unsigned long long)(const void*)w16tblk,
                            (unsigned)(unsigned long long)(uintptr_t)WldsT,
                            NB, DEPTH, NB, /*pad_ivl=*/3, /*pad_amt=*/3);
                __builtin_amdgcn_s_wait_tensorcnt(0);
            }
            manual_stage = false;
        }
#endif
        if (manual_stage) {
            #pragma unroll
            for (int j = 0; j < 8; ++j) {
                const int q = j * THREADS + tid;       // 0..4095 chunks of 8 bf16
                uint4 pk;
                if (use_bf16_src) {
                    pk = *(const uint4*)(w16blk + (size_t)q * 4);
                } else {
                    const int flat = q * 8;
                    const float4 a = *(const float4*)(wb + flat);
                    const float4 b = *(const float4*)(wb + flat + 4);
                    pk.x = pack_bf16x2(a.x, a.y); pk.y = pack_bf16x2(a.z, a.w);
                    pk.z = pack_bf16x2(b.x, b.y); pk.w = pack_bf16x2(b.z, b.w);
                }
                {
                    const int flat = q * 8;
                    *(uint4*)&Wlds[widx(flat >> 10, flat & 1023)] = pk;
                }
                const int d  = q >> 2;
                const int n0 = (q & 3) * 8;
                uint4 pt;
                if (use_bf16_src) {
                    pt = *(const uint4*)(w16tblk + (size_t)q * 4);
                } else {
                    float s[8];
                    #pragma unroll
                    for (int i = 0; i < 8; ++i)
                        s[i] = wb[(size_t)(n0 + i) * DEPTH + d];
                    pt.x = pack_bf16x2(s[0], s[1]); pt.y = pack_bf16x2(s[2], s[3]);
                    pt.z = pack_bf16x2(s[4], s[5]); pt.w = pack_bf16x2(s[6], s[7]);
                }
                *(uint4*)&WldsT[d * TPITCH + n0] = pt;
            }
        }

        // ---- zero this iteration's score buffer ----
        const int buf = nb & 1;
        {
            float* sb = Sfull + buf * (2 * 16 * SPITCH);
            for (int z = tid; z < 2 * 16 * SPITCH; z += THREADS) sb[z] = 0.0f;
        }
        __syncthreads();          // (B) LDS staged, score buffer zeroed

        // ---- partial scores over this wave's 64-col depth slice ----
        #pragma unroll
        for (int nt = 0; nt < 2; ++nt) {
            v8f sacc[2];
            #pragma unroll
            for (int rt = 0; rt < 2; ++rt)
                #pragma unroll
                for (int v = 0; v < 8; ++v) sacc[rt][v] = 0.0f;

            const int row = nt * 16 + ln16;
            #pragma unroll
            for (int c = 0; c < 2; ++c) {
                const int k0 = dw + 32 * c;
                const unsigned short* p0 = &Wlds[widx(row, k0 + koff)];
                const unsigned short* p1 = &Wlds[widx(row, k0 + 16 + koff)];
                v16bf bfr;                             // B lane = n, K along depth
                #pragma unroll
                for (int i = 0; i < 8; ++i) bfr[i]     = bits_to_bf16(p0[i]);
                #pragma unroll
                for (int i = 0; i < 8; ++i) bfr[8 + i] = bits_to_bf16(p1[i]);
                #pragma unroll
                for (int rt = 0; rt < 2; ++rt)
                    sacc[rt] = __builtin_amdgcn_wmma_f32_16x16x32_bf16(
                        false, af[rt][c], false, bfr, (short)0, sacc[rt], false, false);
            }
            #pragma unroll
            for (int rt = 0; rt < 2; ++rt)
                #pragma unroll
                for (int v = 0; v < 8; ++v)
                    atomicAdd(&Sfull[((buf * 2 + rt) * 16 + (v + koff)) * SPITCH
                                     + nt * 16 + ln16],
                              sacc[rt][v]);
        }
        __syncthreads();          // (C) full scores ready

        // ---- online softmax + P fragments (identical math in every wave) ----
        v16bf pfrag[2];
        float scale[2];
        #pragma unroll
        for (int rt = 0; rt < 2; ++rt) {
            const float* srow = &Sfull[((buf * 2 + rt) * 16 + ln16) * SPITCH];
            float sv[16];
            #pragma unroll
            for (int j = 0; j < 8; ++j) sv[j]     = srow[koff + j];
            #pragma unroll
            for (int j = 0; j < 8; ++j) sv[8 + j] = srow[16 + koff + j];

            float mx = sv[0];
            #pragma unroll
            for (int j = 1; j < 16; ++j) mx = fmaxf(mx, sv[j]);
            mx = fmaxf(mx, __shfl_xor(mx, 16, 32));
            const float mnew = fmaxf(mrow[rt], mx);
            scale[rt] = __expf(mrow[rt] - mnew);

            float psum = 0.0f;
            #pragma unroll
            for (int j = 0; j < 16; ++j) {
                const float p = __expf(sv[j] - mnew);
                psum += p;
                pfrag[rt][j] = bits_to_bf16(f32_to_bf16_bits(p));
            }
            psum += __shfl_xor(psum, 16, 32);
            lrow[rt] = lrow[rt] * scale[rt] + psum;
            mrow[rt] = mnew;
        }

        #pragma unroll
        for (int rt = 0; rt < 2; ++rt)
            #pragma unroll
            for (int v = 0; v < 8; ++v) {
                const float sc = __shfl(scale[rt], v + koff, 32);
                #pragma unroll
                for (int t = 0; t < 4; ++t) accO[rt][t][v] *= sc;
            }

        // ---- retrieval: O_tile += P (16x32) @ Wb (32 x 16 d-cols) ----
        #pragma unroll
        for (int t = 0; t < 4; ++t) {
            const unsigned short* wc = &WldsT[(dw + t * 16 + ln16) * TPITCH];
            v16bf bfr;                                 // B lane = d col, K = n index
            #pragma unroll
            for (int j = 0; j < 8; ++j) bfr[j]     = bits_to_bf16(wc[koff + j]);
            #pragma unroll
            for (int j = 0; j < 8; ++j) bfr[8 + j] = bits_to_bf16(wc[16 + koff + j]);
            #pragma unroll
            for (int rt = 0; rt < 2; ++rt)
                accO[rt][t] = __builtin_amdgcn_wmma_f32_16x16x32_bf16(
                    false, pfrag[rt], false, bfr, (short)0, accO[rt][t], false, false);
        }
    }

    // ---- final normalize and store ----
    #pragma unroll
    for (int rt = 0; rt < 2; ++rt) {
        const float inv = 1.0f / lrow[rt];
        #pragma unroll
        for (int v = 0; v < 8; ++v) {
            const float iv = __shfl(inv, v + koff, 32);
            const int m = v + koff;
            float* orow = out + (size_t)(rowbase + rt * 16 + m) * DEPTH + dw;
            #pragma unroll
            for (int t = 0; t < 4; ++t)
                orow[t * 16 + ln16] = accO[rt][t][v] * iv;
        }
    }
}

extern "C" void kernel_launch(void* const* d_in, const int* in_sizes, int n_in,
                              void* d_out, int out_size, void* d_ws, size_t ws_size,
                              hipStream_t stream) {
    (void)in_sizes; (void)n_in; (void)out_size;
    const float* x = (const float*)d_in[0];
    const float* W = (const float*)d_in[1];
    float* out = (float*)d_out;

    const size_t need = (size_t)NMEM * DEPTH * 2 * 2;   // W16 + W16T, 64 MiB
    const int use_bf16 = (d_ws != nullptr && ws_size >= need) ? 1 : 0;
    unsigned int* W16  = (unsigned int*)d_ws;
    unsigned int* W16T = W16 + (size_t)NMEM * DEPTH / 2;

    if (use_bf16) {
        const int pairs = NMEM * DEPTH / 2;             // 8,388,608
        hipLaunchKernelGGL(convert_w_bf16, dim3(pairs / 256), dim3(256), 0, stream,
                           W, W16, W16T);
    }
    hipLaunchKernelGGL(hopfield_flash_wmma, dim3(BATCH / ROWS), dim3(THREADS),
                       SMEM_BYTES, stream, x, W, W16, W16T, out, use_bf16);
}
